// DirectVoxGO_6820408066377
// MI455X (gfx1250) — matrix operand order repaired and checked by reading, not compiled, not run on hardware
//
#include <hip/hip_runtime.h>

typedef _Float16 v16h __attribute__((ext_vector_type(16)));
typedef _Float16 v8h  __attribute__((ext_vector_type(8)));
typedef float    v8f  __attribute__((ext_vector_type(8)));

#define GR   160
#define GR2  25600
#define GVOL 4096000
#define WAVES 4
#define BLOCK 128

// ws layout in halves: fragment-packed f16 weights for WMMA B operands.
// Fragment layout used here (contiguous k order, same as A; wave32, 16-bit B 32x16):
//   lane L: n = L%16, g = L/16; halves idx 0..15 -> K = kt*32 + g*16 + idx
#define W0F_OFF 0      // 2 ktiles * 8 ntiles * 512 halves
#define W1F_OFF 8192   // 4 ktiles * 8 ntiles * 512 halves
#define W2F_OFF 24576  // 4 ktiles * 1 ntile  * 512 halves
#define WS_HALVES 26624

__global__ void dvgo_pack_weights(const float* __restrict__ w0,
                                  const float* __restrict__ w1,
                                  const float* __restrict__ w2,
                                  _Float16* __restrict__ wsf) {
  int t = blockIdx.x * blockDim.x + threadIdx.x;
  if (t >= WS_HALVES) return;
  float v;
  if (t < W1F_OFF) {                       // w0: [36][128], K padded to 64
    int f = t >> 9, r = t & 511;
    int L = r >> 4, i = r & 15;
    int kt = f >> 3, nt = f & 7;
    int K = kt * 32 + (L >> 4) * 16 + i;
    int N = nt * 16 + (L & 15);
    v = (K < 36) ? w0[K * 128 + N] : 0.0f;
  } else if (t < W2F_OFF) {                // w1: [128][128]
    int u = t - W1F_OFF;
    int f = u >> 9, r = u & 511;
    int L = r >> 4, i = r & 15;
    int kt = f >> 3, nt = f & 7;
    int K = kt * 32 + (L >> 4) * 16 + i;
    int N = nt * 16 + (L & 15);
    v = w1[K * 128 + N];
  } else {                                 // w2: [128][3], N padded to 16
    int u = t - W2F_OFF;
    int kt = u >> 9, r = u & 511;
    int L = r >> 4, i = r & 15;
    int K = kt * 32 + (L >> 4) * 16 + i;
    int N = L & 15;
    v = (N < 3) ? w2[K * 3 + N] : 0.0f;
  }
  wsf[t] = (_Float16)v;
}

__device__ __forceinline__ v16h cat16(v8h lo, v8h hi) {
  v16h r;
#pragma unroll
  for (int i = 0; i < 8; ++i) { r[i] = lo[i]; r[i + 8] = hi[i]; }
  return r;
}

// A fragment (16x32 f16) from an LDS row-major [32][128]-half activation
// buffer.  Layout: lane L (m=L%16, g=L/16): halves 0..15 = K kt*32+g*16+0..15
// (matches packed B)  -> two contiguous ds_load_b128.
__device__ __forceinline__ v16h load_a(const _Float16* row, int kt, int g) {
  const v8h* p0 = (const v8h*)(row + kt * 32 + g * 16);
  const v8h* p1 = (const v8h*)(row + kt * 32 + g * 16 + 8);
  return cat16(*p0, *p1);
}

// B fragment: 32 contiguous bytes per lane from the fragment-packed ws.
__device__ __forceinline__ v16h load_b(const _Float16* frag, int lane) {
  const v8h* p = (const v8h*)(frag + lane * 16);
  return cat16(p[0], p[1]);
}

#define WMMA_F16(A, B, C) \
  __builtin_amdgcn_wmma_f32_16x16x32_f16(false, (A), false, (B), (short)0, (C), false, false)

__global__ void __launch_bounds__(BLOCK)
dvgo_main(const float* __restrict__ pts, const float* __restrict__ vds,
          const float* __restrict__ dgrid, const float* __restrict__ cgrid,
          const float* __restrict__ b0, const float* __restrict__ b1,
          const float* __restrict__ b2, const _Float16* __restrict__ wsf,
          float* __restrict__ out, int npts, int ntiles) {
  // Per-wave ping/pong activation buffers: [32 rows][128 halves].
  // bufA: feat (cols 0..63) then reused for h2; bufB: h.  4 waves -> 64 KB.
  __shared__ __align__(16) _Float16 smem[WAVES][2][32 * 128];
  const int wave = threadIdx.x >> 5;
  const int lane = threadIdx.x & 31;
  const int g = lane >> 4;   // lane half-group
  const int m = lane & 15;   // row (A) / col (B,C,D) within tile

  int tile = blockIdx.x * WAVES + wave;
  if (tile >= ntiles) tile = ntiles - 1;  // duplicate last tile; keeps barriers uniform
  const int pbase = tile * 32;
  const int p = pbase + lane;
  const int pc = (p < npts) ? p : (npts - 1);

  _Float16* bufA = smem[wave][0];
  _Float16* bufB = smem[wave][1];

  // ---------------- Phase 1: trilerp + view embedding (per-lane point) -----
  float fv[36];
  float dif[3];
  {
    float px = pts[pc * 3 + 0], py = pts[pc * 3 + 1], pz = pts[pc * 3 + 2];
    float ux = fminf(fmaxf((px + 1.0f) * 79.5f, 0.0f), 159.0f);
    float uy = fminf(fmaxf((py + 1.0f) * 79.5f, 0.0f), 159.0f);
    float uz = fminf(fmaxf((pz + 1.0f) * 79.5f, 0.0f), 159.0f);
    int ix = (int)ux; if (ix > 158) ix = 158;
    int iy = (int)uy; if (iy > 158) iy = 158;
    int iz = (int)uz; if (iz > 158) iz = 158;
    float fx = ux - (float)ix, fy = uy - (float)iy, fz = uz - (float)iz;
    float wx[2] = {1.0f - fx, fx};
    float wy[2] = {1.0f - fy, fy};
    float wz[2] = {1.0f - fz, fz};
    int offs[8];
    float wgt[8];
#pragma unroll
    for (int dx = 0; dx < 2; ++dx)
#pragma unroll
      for (int dy = 0; dy < 2; ++dy)
#pragma unroll
        for (int dz = 0; dz < 2; ++dz) {
          int k = dx * 4 + dy * 2 + dz;
          offs[k] = (ix + dx) * GR2 + (iy + dy) * GR + (iz + dz);
          wgt[k] = wx[dx] * wy[dy] * wz[dz];
        }
    float dsum = 0.0f;
#pragma unroll
    for (int k = 0; k < 8; ++k) dsum += wgt[k] * dgrid[offs[k]];
    if (p < npts) out[p * 4 + 3] = dsum;  // density channel

#pragma unroll
    for (int c = 0; c < 12; ++c) {
      const float* gp = cgrid + c * GVOL;
      float s = 0.0f;
#pragma unroll
      for (int k = 0; k < 8; ++k) s += wgt[k] * gp[offs[k]];
      if (c < 3) dif[c] = s; else fv[c - 3] = s;
    }
    float vx = vds[pc * 3 + 0], vy = vds[pc * 3 + 1], vz = vds[pc * 3 + 2];
    fv[9] = vx; fv[10] = vy; fv[11] = vz;
#pragma unroll
    for (int f = 0; f < 4; ++f) {
      float s = (float)(1 << f);
      fv[12 + f * 3 + 0] = __sinf(s * vx);
      fv[12 + f * 3 + 1] = __sinf(s * vy);
      fv[12 + f * 3 + 2] = __sinf(s * vz);
      fv[24 + f * 3 + 0] = __cosf(s * vx);
      fv[24 + f * 3 + 1] = __cosf(s * vy);
      fv[24 + f * 3 + 2] = __cosf(s * vz);
    }
  }
  // Diffuse component this lane will supply to the output transpose later.
  float dsel = (m == 0) ? dif[0] : ((m == 1) ? dif[1] : dif[2]);

  // Stage feat row (f16, K padded 36..63 with zeros) into LDS.
  {
    _Float16* row = bufA + lane * 128;
#pragma unroll
    for (int k = 0; k < 36; ++k) row[k] = (_Float16)fv[k];
#pragma unroll
    for (int k = 36; k < 64; ++k) row[k] = (_Float16)0.0f;
  }
  __syncthreads();

  // ---------------- GEMM1: feat[32x64] x w0[64x128] -> h (bufB), ReLU ------
  for (int nt = 0; nt < 8; ++nt) {
    float bias = b0[nt * 16 + m];
    v8f c0, c1;
#pragma unroll
    for (int j = 0; j < 8; ++j) { c0[j] = bias; c1[j] = bias; }
#pragma unroll
    for (int kt = 0; kt < 2; ++kt) {
      v16h b  = load_b(wsf + W0F_OFF + (kt * 8 + nt) * 512, lane);
      v16h a0 = load_a(bufA + m * 128, kt, g);
      v16h a1 = load_a(bufA + (16 + m) * 128, kt, g);
      c0 = WMMA_F16(a0, b, c0);
      c1 = WMMA_F16(a1, b, c1);
    }
#pragma unroll
    for (int j = 0; j < 8; ++j) {
      bufB[(8 * g + j) * 128 + nt * 16 + m]      = (_Float16)fmaxf(c0[j], 0.0f);
      bufB[(16 + 8 * g + j) * 128 + nt * 16 + m] = (_Float16)fmaxf(c1[j], 0.0f);
    }
  }
  __syncthreads();

  // ---------------- GEMM2: h[32x128] x w1[128x128] -> h2 (bufA), ReLU ------
  for (int nt = 0; nt < 8; ++nt) {
    float bias = b1[nt * 16 + m];
    v8f c0, c1;
#pragma unroll
    for (int j = 0; j < 8; ++j) { c0[j] = bias; c1[j] = bias; }
#pragma unroll
    for (int kt = 0; kt < 4; ++kt) {
      v16h b  = load_b(wsf + W1F_OFF + (kt * 8 + nt) * 512, lane);
      v16h a0 = load_a(bufB + m * 128, kt, g);
      v16h a1 = load_a(bufB + (16 + m) * 128, kt, g);
      c0 = WMMA_F16(a0, b, c0);
      c1 = WMMA_F16(a1, b, c1);
    }
#pragma unroll
    for (int j = 0; j < 8; ++j) {
      bufA[(8 * g + j) * 128 + nt * 16 + m]      = (_Float16)fmaxf(c0[j], 0.0f);
      bufA[(16 + 8 * g + j) * 128 + nt * 16 + m] = (_Float16)fmaxf(c1[j], 0.0f);
    }
  }
  __syncthreads();

  // ---------------- GEMM3: h2[32x128] x w2[128x16(3)] -> logits ------------
  {
    float bias2 = (m < 3) ? b2[m] : 0.0f;
    v8f cs[2];
#pragma unroll
    for (int j = 0; j < 8; ++j) { cs[0][j] = bias2; cs[1][j] = bias2; }
#pragma unroll
    for (int kt = 0; kt < 4; ++kt) {
      v16h b  = load_b(wsf + W2F_OFF + kt * 512, lane);
      v16h a0 = load_a(bufA + m * 128, kt, g);
      v16h a1 = load_a(bufA + (16 + m) * 128, kt, g);
      cs[0] = WMMA_F16(a0, b, cs[0]);
      cs[1] = WMMA_F16(a1, b, cs[1]);
    }
    // sigmoid(logit + diffuse); diffuse transposed in via ds_bpermute (shfl).
#pragma unroll
    for (int tt = 0; tt < 2; ++tt) {
#pragma unroll
      for (int j = 0; j < 8; ++j) {
        int r = tt * 16 + 8 * g + j;           // point row within wave tile
        float diff = __shfl(dsel, r, 32);      // all lanes participate
        float x = cs[tt][j] + diff;
        float sg = 1.0f / (1.0f + __expf(-x));
        int pt = pbase + r;
        if (m < 3 && pt < npts) out[pt * 4 + m] = sg;
      }
    }
  }
}

extern "C" void kernel_launch(void* const* d_in, const int* in_sizes, int n_in,
                              void* d_out, int out_size, void* d_ws, size_t ws_size,
                              hipStream_t stream) {
  const float* pts = (const float*)d_in[0];
  const float* vds = (const float*)d_in[1];
  const float* dg  = (const float*)d_in[2];
  const float* cg  = (const float*)d_in[3];
  const float* w0  = (const float*)d_in[4];
  const float* b0  = (const float*)d_in[5];
  const float* w1  = (const float*)d_in[6];
  const float* b1  = (const float*)d_in[7];
  const float* w2  = (const float*)d_in[8];
  const float* b2  = (const float*)d_in[9];
  float* out = (float*)d_out;
  _Float16* wsf = (_Float16*)d_ws;

  int npts = in_sizes[0] / 3;
  int ntiles = (npts + 31) / 32;
  int nblocks = (ntiles + WAVES - 1) / WAVES;

  dvgo_pack_weights<<<(WS_HALVES + 255) / 256, 256, 0, stream>>>(w0, w1, w2, wsf);
  dvgo_main<<<nblocks, BLOCK, 0, stream>>>(pts, vds, dg, cg, b0, b1, b2, wsf,
                                           out, npts, ntiles);
}
